// LlamaAttention_38319698215439
// MI455X (gfx1250) — matrix operand order repaired and checked by reading, not compiled, not run on hardware
//
#include <hip/hip_runtime.h>

// ---------------------------------------------------------------------------
// LlamaAttention forward for MI455X (gfx1250, wave32, WMMA bf16)
// B=4, S=2048, H=576, 9 heads x 64, 3 KV heads, RoPE base 10000, causal SDPA
// ---------------------------------------------------------------------------

typedef __attribute__((ext_vector_type(16))) __bf16 v16bf;
typedef __attribute__((ext_vector_type(8)))  __bf16 bf16x8;
typedef __attribute__((ext_vector_type(4)))  __bf16 bf16x4;
typedef __attribute__((ext_vector_type(8)))  float  v8f;

#define B_   4
#define S_   2048
#define H_   576
#define NH   9
#define NKV  3
#define HD   64
// ln(10000)/32  (inv_freq(d) = exp(-d * ln(10000)/32), d in [0,32))
#define ROPE_C 0.28782313662425572f
#define LDSTR 40   // LDS row stride in bf16 elems (32 data + 8 pad, keeps 16B align)

__device__ __forceinline__ v16bf make_frag(bf16x8 lo, bf16x8 hi) {
    union { v16bf v; bf16x8 h[2]; } u;
    u.h[0] = lo; u.h[1] = hi;
    return u.v;
}

__device__ __forceinline__ bf16x4 cvt4(float4 f) {
    bf16x4 r;
    r.x = (__bf16)f.x; r.y = (__bf16)f.y;
    r.z = (__bf16)f.z; r.w = (__bf16)f.w;
    return r;
}

__device__ __forceinline__ v8f wmma_bf16(v16bf a, v16bf b, v8f c) {
    return __builtin_amdgcn_wmma_f32_16x16x32_bf16(false, a, false, b,
                                                   (short)0, c, false, false);
}

// 64x64 tile of Y = X @ W^T  (X: [M,576] f32 row-major, W: [N,576] f32 row-major)
// 128 threads = 4 waves; wave w produces rows [m0+16w, m0+16w+16) x cols [n0, n0+64)
// acc[nt][v]: row M = v + 8*(lane>>4) (+16w), col N = nt*16 + (lane&15)
__device__ __forceinline__ void gemm_tile_576(const float* __restrict__ X,
                                              const float* __restrict__ W,
                                              int m0, int n0,
                                              __bf16* Xs, __bf16* Ws, v8f acc[4]) {
    const int t    = threadIdx.x;
    const int lane = t & 31;
    const int w    = t >> 5;
    const int hrow = lane & 15;
    const int hsel = lane >> 4;
#pragma unroll
    for (int i = 0; i < 4; ++i) acc[i] = v8f{};

    // staging slice owned by this thread: row rr (+0/16/32/48), cols cc..cc+3
    const int rr = t >> 3;          // 0..15
    const int cc = (t & 7) * 4;     // 0,4,...,28
    const float* xg = X + (size_t)(m0 + rr) * H_ + cc;
    const float* wg = W + (size_t)(n0 + rr) * H_ + cc;
    const int arow = w * 16 + hrow;

#pragma unroll 1
    for (int k0 = 0; k0 < H_; k0 += 32) {
        // issue all 8 16-byte loads before the barrier (latency overlaps wait)
        float4 xv0 = *(const float4*)(xg + k0);
        float4 xv1 = *(const float4*)(xg + k0 + 16 * H_);
        float4 xv2 = *(const float4*)(xg + k0 + 32 * H_);
        float4 xv3 = *(const float4*)(xg + k0 + 48 * H_);
        float4 wv0 = *(const float4*)(wg + k0);
        float4 wv1 = *(const float4*)(wg + k0 + 16 * H_);
        float4 wv2 = *(const float4*)(wg + k0 + 32 * H_);
        float4 wv3 = *(const float4*)(wg + k0 + 48 * H_);

        __syncthreads();   // previous iteration's fragment reads complete
        *(bf16x4*)&Xs[(rr +  0) * LDSTR + cc] = cvt4(xv0);
        *(bf16x4*)&Xs[(rr + 16) * LDSTR + cc] = cvt4(xv1);
        *(bf16x4*)&Xs[(rr + 32) * LDSTR + cc] = cvt4(xv2);
        *(bf16x4*)&Xs[(rr + 48) * LDSTR + cc] = cvt4(xv3);
        *(bf16x4*)&Ws[(rr +  0) * LDSTR + cc] = cvt4(wv0);
        *(bf16x4*)&Ws[(rr + 16) * LDSTR + cc] = cvt4(wv1);
        *(bf16x4*)&Ws[(rr + 32) * LDSTR + cc] = cvt4(wv2);
        *(bf16x4*)&Ws[(rr + 48) * LDSTR + cc] = cvt4(wv3);
        __syncthreads();

        // A fragment (16x32): lane holds row (lane&15); K = hsel*16 + j
        v16bf a = make_frag(*(const bf16x8*)&Xs[arow * LDSTR + hsel * 16],
                            *(const bf16x8*)&Xs[arow * LDSTR + hsel * 16 + 8]);
#pragma unroll
        for (int nt = 0; nt < 4; ++nt) {
            // B fragment (32x16): lane holds col (lane&15); K = hsel*16 + j
            const int bcol = nt * 16 + hrow;
            v16bf b = make_frag(*(const bf16x8*)&Ws[bcol * LDSTR + hsel * 16],
                                *(const bf16x8*)&Ws[bcol * LDSTR + hsel * 16 + 8]);
            acc[nt] = wmma_bf16(a, b, acc[nt]);
        }
    }
}

// ---------------------------------------------------------------------------
// Kernel 1: fused QKV projection + RoPE.
// grid.x = 128 (row tiles of 64 over B*S=8192), grid.y = 15 (9 Q + 3 K + 3 V heads)
// Q -> qws bf16 [b][h][s][d], K -> kws bf16 [b][kh][s][d], V -> vtws bf16 [b][kh][d][s]
// ---------------------------------------------------------------------------
__global__ void qkv_rope_kernel(const float* __restrict__ hs,
                                const float* __restrict__ Wq,
                                const float* __restrict__ Wk,
                                const float* __restrict__ Wv,
                                __bf16* __restrict__ qws,
                                __bf16* __restrict__ kws,
                                __bf16* __restrict__ vtws) {
    __shared__ __align__(16) __bf16 Xs[64 * LDSTR];
    __shared__ __align__(16) __bf16 Ws[64 * LDSTR];

    const int mt = blockIdx.x;
    const int ct = blockIdx.y;
    const float* W;
    int head, mode;  // 0=Q(rope) 1=K(rope) 2=V(transposed)
    if (ct < 9)       { W = Wq; head = ct;      mode = 0; }
    else if (ct < 12) { W = Wk; head = ct - 9;  mode = 1; }
    else              { W = Wv; head = ct - 12; mode = 2; }

    v8f acc[4];
    gemm_tile_576(hs, W, mt * 64, head * 64, Xs, Ws, acc);

    const int t = threadIdx.x, lane = t & 31, w = t >> 5;
    const int hrow = lane & 15, hsel = lane >> 4;

    if (mode == 2) {
#pragma unroll
        for (int nt = 0; nt < 4; ++nt) {
            const int d = nt * 16 + hrow;
#pragma unroll
            for (int v = 0; v < 8; ++v) {
                const int row = mt * 64 + w * 16 + v + 8 * hsel;
                const int b = row >> 11, s = row & (S_ - 1);
                vtws[(size_t)((b * NKV + head) * HD + d) * S_ + s] = (__bf16)acc[nt][v];
            }
        }
    } else {
        __bf16* dst = (mode == 0) ? qws : kws;
        const int nh = (mode == 0) ? NH : NKV;
#pragma unroll
        for (int nt = 0; nt < 2; ++nt) {
            const int dp = nt * 16 + hrow;                 // 0..31
            const float invf = __expf(-(float)dp * ROPE_C);
#pragma unroll
            for (int v = 0; v < 8; ++v) {
                const int row = mt * 64 + w * 16 + v + 8 * hsel;
                const int b = row >> 11, s = row & (S_ - 1);
                const float ang = (float)s * invf;
                const float c = __cosf(ang), sn = __sinf(ang);
                const float x1 = acc[nt][v], x2 = acc[nt + 2][v];
                const size_t base = (size_t)((b * nh + head) * S_ + s) * HD;
                dst[base + dp]      = (__bf16)(x1 * c - x2 * sn);
                dst[base + dp + 32] = (__bf16)(x2 * c + x1 * sn);
            }
        }
    }
}

// ---------------------------------------------------------------------------
// Kernel 2: causal flash attention, online softmax.
// grid = (S/64, NH, B), 128 threads; each wave owns 16 query rows x full head.
// ---------------------------------------------------------------------------
__global__ void attn_kernel(const __bf16* __restrict__ qws,
                            const __bf16* __restrict__ kws,
                            const __bf16* __restrict__ vtws,
                            float* __restrict__ attn_out) {
    __shared__ __align__(16) __bf16 Ps[4][16 * LDSTR];   // per-wave P transpose buffer

    const int t = threadIdx.x, lane = t & 31, w = t >> 5;
    const int hrow = lane & 15, hsel = lane >> 4;
    const int q0 = blockIdx.x * 64 + w * 16;
    const int h = blockIdx.y, b = blockIdx.z;
    const int kvh = h / (NH / NKV);

    const __bf16* qp = qws  + (size_t)((b * NH  + h)   * S_ + q0) * HD;
    const __bf16* kp = kws  + (size_t)((b * NKV + kvh) * S_) * HD;
    const __bf16* vp = vtws + (size_t)((b * NKV + kvh) * HD) * S_;

    // Q A-fragments for head-dim K=0..31 and K=32..63
    const __bf16* qr = qp + hrow * HD;
    const v16bf qa0 = make_frag(*(const bf16x8*)(qr + hsel * 16),
                                *(const bf16x8*)(qr + 8 + hsel * 16));
    const v16bf qa1 = make_frag(*(const bf16x8*)(qr + 32 + hsel * 16),
                                *(const bf16x8*)(qr + 40 + hsel * 16));

    v8f o[4];
#pragma unroll
    for (int i = 0; i < 4; ++i) o[i] = v8f{};
    float m[8], l[8];
#pragma unroll
    for (int v = 0; v < 8; ++v) { m[v] = -__builtin_inff(); l[v] = 0.f; }

    const float scale = 0.125f;  // 1/sqrt(64)
    __bf16* ps = &Ps[w][0];

#pragma unroll 1
    for (int kv0 = 0; kv0 <= q0 + 15; kv0 += 32) {
        // ---- scores: two 16x16 tiles covering kv0..kv0+31
        v8f s0 = v8f{}, s1 = v8f{};
        const __bf16* kr0 = kp + (size_t)(kv0 + hrow) * HD + hsel * 16;
        s0 = wmma_bf16(qa0, make_frag(*(const bf16x8*)(kr0),      *(const bf16x8*)(kr0 + 8)),  s0);
        s0 = wmma_bf16(qa1, make_frag(*(const bf16x8*)(kr0 + 32), *(const bf16x8*)(kr0 + 40)), s0);
        const __bf16* kr1 = kr0 + 16 * HD;
        s1 = wmma_bf16(qa0, make_frag(*(const bf16x8*)(kr1),      *(const bf16x8*)(kr1 + 8)),  s1);
        s1 = wmma_bf16(qa1, make_frag(*(const bf16x8*)(kr1 + 32), *(const bf16x8*)(kr1 + 40)), s1);

        // ---- online softmax over the 32 kv columns
#pragma unroll
        for (int v = 0; v < 8; ++v) {
            const int qrow = q0 + v + 8 * hsel;
            const int ka = kv0 + hrow;
            const float v0 = (ka      <= qrow) ? s0[v] * scale : -1e30f;
            const float v1 = (ka + 16 <= qrow) ? s1[v] * scale : -1e30f;
            float rmax = fmaxf(v0, v1);
#pragma unroll
            for (int off = 8; off; off >>= 1) rmax = fmaxf(rmax, __shfl_xor(rmax, off));
            const float mn = fmaxf(m[v], rmax);
            const float corr = __expf(m[v] - mn);
            m[v] = mn;
            const float p0 = __expf(v0 - mn);
            const float p1 = __expf(v1 - mn);
            float rs = p0 + p1;
#pragma unroll
            for (int off = 8; off; off >>= 1) rs += __shfl_xor(rs, off);
            l[v] = l[v] * corr + rs;
#pragma unroll
            for (int nt = 0; nt < 4; ++nt) o[nt][v] *= corr;
            const int prow = v + 8 * hsel;
            ps[prow * LDSTR + hrow]      = (__bf16)p0;
            ps[prow * LDSTR + 16 + hrow] = (__bf16)p1;
        }

        // ---- P (16x32, via same-wave LDS transpose) x V (32x64)
        const v16bf pa = make_frag(*(const bf16x8*)&ps[hrow * LDSTR + hsel * 16],
                                   *(const bf16x8*)&ps[hrow * LDSTR + hsel * 16 + 8]);
#pragma unroll
        for (int nt = 0; nt < 4; ++nt) {
            const int d = nt * 16 + hrow;
            const __bf16* vr = vp + (size_t)d * S_ + kv0 + hsel * 16;
            o[nt] = wmma_bf16(pa, make_frag(*(const bf16x8*)vr, *(const bf16x8*)(vr + 8)), o[nt]);
        }
    }

    // ---- normalize + store fp32 [B,S,H]
#pragma unroll
    for (int v = 0; v < 8; ++v) {
        const float inv = 1.0f / l[v];
        const int s = q0 + v + 8 * hsel;
#pragma unroll
        for (int nt = 0; nt < 4; ++nt) {
            attn_out[((size_t)(b * S_ + s)) * H_ + h * HD + nt * 16 + hrow] = o[nt][v] * inv;
        }
    }
}

// ---------------------------------------------------------------------------
// Kernel 3: output projection  out = attn @ Wo^T   (fp32 store)
// ---------------------------------------------------------------------------
__global__ void oproj_kernel(const float* __restrict__ attn,
                             const float* __restrict__ Wo,
                             float* __restrict__ out) {
    __shared__ __align__(16) __bf16 Xs[64 * LDSTR];
    __shared__ __align__(16) __bf16 Ws[64 * LDSTR];

    const int mt = blockIdx.x;
    const int ct = blockIdx.y;
    v8f acc[4];
    gemm_tile_576(attn, Wo, mt * 64, ct * 64, Xs, Ws, acc);

    const int t = threadIdx.x, lane = t & 31, w = t >> 5;
    const int hrow = lane & 15, hsel = lane >> 4;
#pragma unroll
    for (int nt = 0; nt < 4; ++nt) {
#pragma unroll
        for (int v = 0; v < 8; ++v) {
            const int row = mt * 64 + w * 16 + v + 8 * hsel;
            out[(size_t)row * H_ + ct * 64 + nt * 16 + hrow] = acc[nt][v];
        }
    }
}

// ---------------------------------------------------------------------------
extern "C" void kernel_launch(void* const* d_in, const int* in_sizes, int n_in,
                              void* d_out, int out_size, void* d_ws, size_t ws_size,
                              hipStream_t stream) {
    const float* hs = (const float*)d_in[0];
    const float* Wq = (const float*)d_in[1];
    const float* Wk = (const float*)d_in[2];
    const float* Wv = (const float*)d_in[3];
    const float* Wo = (const float*)d_in[4];
    float* out = (float*)d_out;

    char* ws = (char*)d_ws;
    const size_t qbytes = (size_t)B_ * NH  * S_ * HD * sizeof(__bf16);  // 9.4 MB
    const size_t kbytes = (size_t)B_ * NKV * S_ * HD * sizeof(__bf16);  // 3.1 MB
    __bf16* qws  = (__bf16*)(ws);
    __bf16* kws  = (__bf16*)(ws + qbytes);
    __bf16* vtws = (__bf16*)(ws + qbytes + kbytes);
    float*  aws  = (float*)(ws + qbytes + 2 * kbytes);                  // 18.9 MB

    dim3 g1(128, 15, 1);               // 64-row tiles x (9 Q + 3 K + 3 V) heads
    qkv_rope_kernel<<<g1, 128, 0, stream>>>(hs, Wq, Wk, Wv, qws, kws, vtws);

    dim3 g2(S_ / 64, NH, B_);
    attn_kernel<<<g2, 128, 0, stream>>>(qws, kws, vtws, aws);

    dim3 g3(128, NH, 1);
    oproj_kernel<<<g3, 128, 0, stream>>>(aws, Wo, out);
}
